// N3AggregationBase_55018531062325
// MI455X (gfx1250) — matrix operand order repaired and checked by reading, not compiled, not run on hardware
//
#include <hip/hip_runtime.h>
#include <hip/hip_bf16.h>

// ---------------- problem constants ----------------
#define QN   65536      // H*W queries
#define ON   32         // candidates per query
#define EN   64         // embedding dim
#define FN   147        // C*PS*PS
#define KN   7          // neighbor volumes
#define CN   3
#define PSN  7
#define HN   256
#define WN   256
#define KC   21         // K*C
#define PP   49         // PS*PS
// z layout: [q][p (49)][kc padded to 24] halves -> per-q stride 1176 halves
#define ZPQ  1176
// K2 LDS: staged patches stored column-major [col (160)][o padded to 40] halves
#define COLS 160
#define OPAD 40

typedef __attribute__((ext_vector_type(16))) _Float16 v16h;
typedef __attribute__((ext_vector_type(8)))  _Float16 v8h;
typedef __attribute__((ext_vector_type(8)))  float    v8f;

__device__ __forceinline__ float wave_sum(float v) {
#pragma unroll
    for (int off = 16; off; off >>= 1) v += __shfl_xor(v, off, 32);
    return v;
}
__device__ __forceinline__ float wave_max(float v) {
#pragma unroll
    for (int off = 16; off; off >>= 1) v = fmaxf(v, __shfl_xor(v, off, 32));
    return v;
}

// =====================================================================
// Kernel 1: distances + avg-pooled log-temp + K rounds of NNN softmax.
// One wave per query; lane o <-> candidate o.
// Output: W transposed as f16 [Q][k (8)][o (32)] -> A-fragment-friendly.
// =====================================================================
__global__ void __launch_bounds__(256)
k1_search_nnn(const float* __restrict__ xe, const float* __restrict__ ye,
              const float* __restrict__ log_temp, const int* __restrict__ cand,
              _Float16* __restrict__ Wbuf) {
    __shared__ __align__(16) float sye[8][EN];
    const int lane = threadIdx.x & 31;
    const int wv   = threadIdx.x >> 5;
    const int q    = blockIdx.x * 8 + wv;

    // stage ye[q] into LDS (same-wave DS ordering suffices)
    sye[wv][lane]      = ye[(size_t)q * EN + lane];
    sye[wv][lane + 32] = ye[(size_t)q * EN + 32 + lane];

    const int idx = cand[(size_t)q * ON + lane];
    const float4* xg  = (const float4*)(xe + (size_t)idx * EN);
    const float4* yev = (const float4*)(&sye[wv][0]);
    float dot = 0.f, xg2 = 0.f;
#pragma unroll
    for (int t = 0; t < EN / 4; ++t) {
        float4 a = yev[t], b = xg[t];
        dot = fmaf(a.x, b.x, fmaf(a.y, b.y, fmaf(a.z, b.z, fmaf(a.w, b.w, dot))));
        xg2 = fmaf(b.x, b.x, fmaf(b.y, b.y, fmaf(b.z, b.z, fmaf(b.w, b.w, xg2))));
    }
    float yp  = sye[wv][lane] * sye[wv][lane] + sye[wv][lane + 32] * sye[wv][lane + 32];
    float ye2 = wave_sum(yp);

    // 7x7 avg-pool of log_temp around pixel (zero padding), /49 always
    const int iy = q >> 8, jx = q & 255;
    float ls = 0.f;
    {
        int e = lane;                         // e < 49 always for lane < 32
        if (e < PP) {
            int dy = e / PSN - 3, dx = e % PSN - 3;
            int yy = iy + dy, xx = jx + dx;
            if ((unsigned)yy < HN && (unsigned)xx < WN) ls += log_temp[yy * WN + xx];
        }
        e = lane + 32;
        if (e < PP) {
            int dy = e / PSN - 3, dx = e % PSN - 3;
            int yy = iy + dy, xx = jx + dx;
            if ((unsigned)yy < HN && (unsigned)xx < WN) ls += log_temp[yy * WN + xx];
        }
    }
    float lt   = wave_sum(ls) * (1.f / 49.f);
    float invT = __expf(-lt);
    float logit = -(ye2 + xg2 - 2.f * dot) * invT;

    // neural-nearest-neighbor weights, K rounds; store [q][k][o] (o coalesced)
    _Float16* wq = Wbuf + (size_t)q * (8 * ON);
#pragma unroll
    for (int k = 0; k < KN; ++k) {
        float m = wave_max(logit);
        float e = __expf(logit - m);
        float s = wave_sum(e);
        float w = e / s;
        wq[k * ON + lane] = (_Float16)w;
        logit += __logf(fmaxf(1.f - w, 1e-6f));
    }
    wq[7 * ON + lane] = (_Float16)0.f;        // pad row (A rows 7..15 unused)
}

// =====================================================================
// Kernel 2: wpsum via WMMA. One wave per query.
//   D[7 x 147] = W^T[7(->16) x 32] * Xp[32 x 147(->10 tiles of 16)]
//   K-dim of v_wmma_f32_16x16x32_f16 == O == 32 exactly.
// LDS patch layout [col][o pad 40]: B fragment = 2x ds_load_b128/lane,
// conflict-free (stride 20 DWORDs tiles all 64 banks).
// A fragment = 2x global b128/lane from the [q][k][o] W layout.
// =====================================================================
__global__ void __launch_bounds__(128)
k2_wpsum_wmma(const float* __restrict__ x, const int* __restrict__ cand,
              const _Float16* __restrict__ Wbuf, _Float16* __restrict__ zbuf) {
    __shared__ __align__(16) _Float16 sxp[4][COLS * OPAD];
    const int lane = threadIdx.x & 31;
    const int wv   = threadIdx.x >> 5;
    const int q    = blockIdx.x * 4 + wv;

    const int cidx = cand[(size_t)q * ON + lane];

    // ---- stage 32 gathered patch rows (f16) into LDS, coalesced per row ----
#pragma unroll 1
    for (int o = 0; o < ON; ++o) {
        int row = __shfl(cidx, o, 32);
        if (o < ON - 1) {
            int nrow = __shfl(cidx, o + 1, 32);
            __builtin_prefetch(x + (size_t)nrow * FN, 0, 3);   // global_prefetch_b8
        }
        const float* xr = x + (size_t)row * FN;
#pragma unroll
        for (int t = 0; t < 5; ++t) {
            int col = t * 32 + lane;                            // covers 0..159 exactly
            sxp[wv][col * OPAD + o] = (col < FN) ? (_Float16)xr[col] : (_Float16)0.f;
        }
    }

    const int n  = lane & 15;
    const int hi = lane >> 4;

    // ---- A fragment: W^T 16x32 f16; half h of VGPR i: K(o) = 2(i&3)+j + hi*8 + (i>=4)*16
    //      With W stored [q][k][o], lane needs o-runs {hi*8..hi*8+7} and {+16}: 2x b128.
    const _Float16* wq = Wbuf + (size_t)q * (8 * ON) + (size_t)(n & 7) * ON + hi * 8;
    v8h a0 = *(const v8h*)(wq);
    v8h a1 = *(const v8h*)(wq + 16);
    v16h a;
#pragma unroll
    for (int h = 0; h < 8; ++h) { a[h] = a0[h]; a[h + 8] = a1[h]; }

    // ---- 10 WMMA tiles over F (unroll 2: pipeline DS loads vs WMMA) ----
#pragma unroll 2
    for (int ft = 0; ft < 10; ++ft) {
        // B fragment 32x16 f16: element s -> K = hi*8 + s (s<8), 16 + hi*8 + (s-8) (s>=8);
        // LDS runs at [col*OPAD + hi*8 .. +7] and [+16 .. +23]: two aligned b128 loads.
        const _Float16* bp = &sxp[wv][(ft * 16 + n) * OPAD + hi * 8];
        v8h b0 = *(const v8h*)(bp);
        v8h b1 = *(const v8h*)(bp + 16);
        v16h b;
#pragma unroll
        for (int h = 0; h < 8; ++h) { b[h] = b0[h]; b[h + 8] = b1[h]; }

        v8f cz = (v8f)0;
        v8f d = __builtin_amdgcn_wmma_f32_16x16x32_f16(
            /*neg_a=*/false, a, /*neg_b=*/false, b,
            /*c_mod=*/(short)0, cz, /*reuse_a=*/false, /*reuse_b=*/false);

        // D layout: VGPR r, lanes 0..15 hold rows M=0..7 -> k = r (rows 7..15 unused)
        if (lane < 16) {
            int f = ft * 16 + n;
            if (f < FN) {
                int c = (f >= 2 * PP) ? 2 : ((f >= PP) ? 1 : 0);
                int p = f - c * PP;
                _Float16* zq = zbuf + (size_t)q * ZPQ + p * 24 + c;
#pragma unroll
                for (int r = 0; r < KN; ++r) zq[r * CN] = (_Float16)d[r];
            }
        }
    }
}

// =====================================================================
// Kernel 3: fold as a gather (qindex == arange). One thread per pixel.
// vid[kc,y,x] = sum over 49 valid neighbor queries of z[q][p][kc];
// wvid is the analytic patch-coverage count. Each z chunk read once.
// =====================================================================
__global__ void __launch_bounds__(256)
k3_fold(const _Float16* __restrict__ zbuf, float* __restrict__ out) {
    const int pix = blockIdx.x * 256 + threadIdx.x;
    const int y = pix >> 8, xx = pix & 255;

    float acc[KC];
#pragma unroll
    for (int t = 0; t < KC; ++t) acc[t] = 0.f;

#pragma unroll 1
    for (int dy = -3; dy <= 3; ++dy) {
        int i = y + dy;
        if ((unsigned)i >= HN) continue;
#pragma unroll
        for (int dx = -3; dx <= 3; ++dx) {
            int j = xx + dx;
            if ((unsigned)j >= WN) continue;
            int q = i * WN + j;
            int p = (3 - dy) * PSN + (3 - dx);
            const v8h* zp = (const v8h*)(zbuf + (size_t)q * ZPQ + p * 24);
            v8h h0 = zp[0], h1 = zp[1], h2 = zp[2];
#pragma unroll
            for (int t = 0; t < 8;  ++t) acc[t]      += (float)h0[t];
#pragma unroll
            for (int t = 0; t < 8;  ++t) acc[t + 8]  += (float)h1[t];
#pragma unroll
            for (int t = 0; t < 5;  ++t) acc[t + 16] += (float)h2[t];
        }
    }

    // vid
#pragma unroll
    for (int kc = 0; kc < KC; ++kc)
        out[(size_t)kc * (HN * WN) + pix] = acc[kc];
    // wvid: analytic coverage count (fold of ones), identical across channels
    int cy = min(y, 3) + min(HN - 1 - y, 3) + 1;
    int cx = min(xx, 3) + min(WN - 1 - xx, 3) + 1;
    float cnt = (float)(cy * cx);
#pragma unroll
    for (int kc = 0; kc < KC; ++kc)
        out[(size_t)(KC + kc) * (HN * WN) + pix] = cnt;
}

// =====================================================================
extern "C" void kernel_launch(void* const* d_in, const int* in_sizes, int n_in,
                              void* d_out, int out_size, void* d_ws, size_t ws_size,
                              hipStream_t stream) {
    const float* x        = (const float*)d_in[0];   // [Q,147]
    const float* xe       = (const float*)d_in[1];   // [Q,64]
    const float* ye       = (const float*)d_in[2];   // [Q,64]
    const float* log_temp = (const float*)d_in[3];   // [1,1,256,256]
    const int*   cand     = (const int*)d_in[4];     // [Q,32]
    // d_in[5] qindex == arange(Q): fold handled as gather in k3
    float* out = (float*)d_out;

    _Float16* Wbuf = (_Float16*)d_ws;                                  // Q*8*32 halves = 32 MB
    _Float16* zbuf = (_Float16*)((char*)d_ws + (size_t)QN * ON * 8 * 2); // Q*1176 halves = 154 MB

    k1_search_nnn<<<QN / 8, 256, 0, stream>>>(xe, ye, log_temp, cand, Wbuf);
    k2_wpsum_wmma<<<QN / 4, 128, 0, stream>>>(x, cand, Wbuf, zbuf);
    k3_fold<<<QN / 256, 256, 0, stream>>>(zbuf, out);
}
